// SingleStandFlaxRwkvBlock_76879914598669
// MI455X (gfx1250) — compile-verified, matrix operand order checked by reading, not run on hardware
//
#include <hip/hip_runtime.h>

#define T_SEQ 4096
#define H_DIM 2048
#define I_DIM 8192
#define LN_EPS 1e-5f
#define LDS_PITCH 80  // bytes per staged B-tile row: 64 data + 16 pad (bank spread)

typedef __attribute__((ext_vector_type(16))) __bf16 v16bf;
typedef __attribute__((ext_vector_type(8)))  __bf16 v8bf;
typedef __attribute__((ext_vector_type(8)))  float  v8f;

// round-to-nearest-even f32 -> bf16 bits
__device__ __forceinline__ unsigned short f32_to_bf16(float f) {
  unsigned int u = __float_as_uint(f);
  u += 0x7fffu + ((u >> 16) & 1u);
  return (unsigned short)(u >> 16);
}

// Wave-relative LDS byte offset of a generic pointer to __shared__ memory
// (the low 32 bits of the generic address).
__device__ __forceinline__ unsigned lds_off(const void* p) {
  return (unsigned)(unsigned long long)p;
}

// async copy: 16 bytes global -> LDS, tracked by ASYNCcnt
__device__ __forceinline__ void async_b128_to_lds(unsigned lds, const void* g) {
  asm volatile("global_load_async_to_lds_b128 %0, %1, off"
               :: "v"(lds), "v"((unsigned long long)g)
               : "memory");
}

// build a 16-bit WMMA fragment from two 16B chunks (CDNA5 16-bit layout:
// lanes 0-15 carry K 0..7 / 16..23, lanes 16-31 carry K 8..15 / 24..31)
__device__ __forceinline__ v16bf frag_from_global(const unsigned short* row,
                                                  int k0, int koff) {
  const v8bf lo = *reinterpret_cast<const v8bf*>(row + k0 + koff);
  const v8bf hi = *reinterpret_cast<const v8bf*>(row + k0 + koff + 16);
  return __builtin_shufflevector(lo, hi,
      0, 1, 2, 3, 4, 5, 6, 7, 8, 9, 10, 11, 12, 13, 14, 15);
}
__device__ __forceinline__ v16bf frag_from_lds(const unsigned char* row,
                                               int koff) {
  const v8bf lo = *reinterpret_cast<const v8bf*>(row + 2 * koff);
  const v8bf hi = *reinterpret_cast<const v8bf*>(row + 32 + 2 * koff);
  return __builtin_shufflevector(lo, hi,
      0, 1, 2, 3, 4, 5, 6, 7, 8, 9, 10, 11, 12, 13, 14, 15);
}

// ---------------------------------------------------------------------------
// Weight prep: Wt[c*R + r] = bf16(W[r*C + c])  (K-contiguous [N][K] layout)
// ---------------------------------------------------------------------------
__global__ void k_convert_transpose(const float* __restrict__ W,
                                    unsigned short* __restrict__ Wt,
                                    int R, int C) {
  long long idx = (long long)blockIdx.x * blockDim.x + threadIdx.x;
  long long total = (long long)R * C;
  if (idx >= total) return;
  int r = (int)(idx / C);
  int c = (int)(idx % C);
  Wt[(long long)c * R + r] = f32_to_bf16(W[idx]);
}

// ---------------------------------------------------------------------------
// Row LayerNorm: one block (256 threads) per row of [T,H]
// ---------------------------------------------------------------------------
__global__ void k_layernorm(const float* __restrict__ x,
                            const float* __restrict__ scale,
                            const float* __restrict__ bias,
                            float* __restrict__ y) {
  __shared__ float red[256];
  const int t = blockIdx.x;
  const float* row = x + (long long)t * H_DIM;
  float s = 0.f;
  for (int i = threadIdx.x; i < H_DIM; i += 256) s += row[i];
  red[threadIdx.x] = s; __syncthreads();
  for (int off = 128; off > 0; off >>= 1) {
    if (threadIdx.x < off) red[threadIdx.x] += red[threadIdx.x + off];
    __syncthreads();
  }
  const float mu = red[0] * (1.f / H_DIM);
  __syncthreads();
  float v = 0.f;
  for (int i = threadIdx.x; i < H_DIM; i += 256) { float d = row[i] - mu; v += d * d; }
  red[threadIdx.x] = v; __syncthreads();
  for (int off = 128; off > 0; off >>= 1) {
    if (threadIdx.x < off) red[threadIdx.x] += red[threadIdx.x + off];
    __syncthreads();
  }
  const float rstd = rsqrtf(red[0] * (1.f / H_DIM) + LN_EPS);
  float* orow = y + (long long)t * H_DIM;
  for (int i = threadIdx.x; i < H_DIM; i += 256)
    orow[i] = (row[i] - mu) * rstd * scale[i] + bias[i];
}

// ---------------------------------------------------------------------------
// Attention token-shift mix -> bf16 kx/vx/rx; also emit x1 last row
// ---------------------------------------------------------------------------
__global__ void k_mix_att(const float* __restrict__ x1,
                          const float* __restrict__ att_sx,
                          const float* __restrict__ tm_key,
                          const float* __restrict__ tm_value,
                          const float* __restrict__ tm_recep,
                          unsigned short* __restrict__ kxb,
                          unsigned short* __restrict__ vxb,
                          unsigned short* __restrict__ rxb,
                          float* __restrict__ x1_last_out) {
  long long idx = (long long)blockIdx.x * blockDim.x + threadIdx.x;
  if (idx >= (long long)T_SEQ * H_DIM) return;
  const int h = (int)(idx % H_DIM);
  const int t = (int)(idx / H_DIM);
  const float xc = x1[idx];
  const float xp = (t == 0) ? att_sx[h] : x1[idx - H_DIM];
  const float mk = tm_key[h], mv = tm_value[h], mr = tm_recep[h];
  kxb[idx] = f32_to_bf16(xc * mk + xp * (1.f - mk));
  vxb[idx] = f32_to_bf16(xc * mv + xp * (1.f - mv));
  rxb[idx] = f32_to_bf16(xc * mr + xp * (1.f - mr));
  if (t == T_SEQ - 1) x1_last_out[h] = xc;
}

// ---------------------------------------------------------------------------
// FFN token-shift mix -> bf16 fk/fr; also emit x2 last row
// ---------------------------------------------------------------------------
__global__ void k_mix_ffn(const float* __restrict__ x2,
                          const float* __restrict__ ffd_sx,
                          const float* __restrict__ ffn_tm_key,
                          const float* __restrict__ ffn_tm_recep,
                          unsigned short* __restrict__ fkb,
                          unsigned short* __restrict__ frb,
                          float* __restrict__ x2_last_out) {
  long long idx = (long long)blockIdx.x * blockDim.x + threadIdx.x;
  if (idx >= (long long)T_SEQ * H_DIM) return;
  const int h = (int)(idx % H_DIM);
  const int t = (int)(idx / H_DIM);
  const float xc = x2[idx];
  const float xp = (t == 0) ? ffd_sx[h] : x2[idx - H_DIM];
  const float mk = ffn_tm_key[h], mr = ffn_tm_recep[h];
  fkb[idx] = f32_to_bf16(xc * mk + xp * (1.f - mk));
  frb[idx] = f32_to_bf16(xc * mr + xp * (1.f - mr));
  if (t == T_SEQ - 1) x2_last_out[h] = xc;
}

// ---------------------------------------------------------------------------
// WKV scan: one lane per channel; coalesced k/v/r reads per timestep.
// ---------------------------------------------------------------------------
__global__ void k_wkv(const float* __restrict__ kbuf,
                      const float* __restrict__ vbuf,
                      const float* __restrict__ rbuf,
                      const float* __restrict__ att_aa,
                      const float* __restrict__ att_bb,
                      const float* __restrict__ att_pp,
                      const float* __restrict__ time_decay,
                      const float* __restrict__ time_first,
                      unsigned short* __restrict__ rcxb,
                      float* __restrict__ aa_out,
                      float* __restrict__ bb_out,
                      float* __restrict__ pp_out) {
  const int h = blockIdx.x * blockDim.x + threadIdx.x;
  if (h >= H_DIM) return;
  float aa = att_aa[h], bb = att_bb[h], pp = att_pp[h];
  const float tf = time_first[h];
  const float wd = -__expf(time_decay[h]);
  for (int t = 0; t < T_SEQ; ++t) {
    const long long idx = (long long)t * H_DIM + h;
    const float kk = kbuf[idx], vv = vbuf[idx];
    const float ww = tf + kk;
    const float p  = fmaxf(pp, ww);
    const float e1 = __expf(pp - p);
    const float e2 = __expf(ww - p);
    const float ct = (e1 * aa + e2 * vv) / (e1 * bb + e2);
    rcxb[idx] = f32_to_bf16(rbuf[idx] * ct);
    const float ww2 = wd + pp;
    const float p2  = fmaxf(ww2, kk);
    const float e1b = __expf(ww2 - p2);
    const float e2b = __expf(kk - p2);
    aa = e1b * aa + e2b * vv;
    bb = e1b * bb + e2b;
    pp = p2;
  }
  aa_out[h] = aa; bb_out[h] = bb; pp_out[h] = pp;
}

// ---------------------------------------------------------------------------
// WMMA bf16 GEMM: C[M,N] = A[M,K] * Bt[N,K]^T, fp32 accumulate.
// Block = 256 threads = 8 waves stacked in M; wave tile 32x64 -> block 256x64.
// Per k-step (K=32): 2 A fragments (register double-buffered, prefetched one
// step ahead) x 4 B fragments (from the async-staged LDS tile, loaded into
// DISTINCT registers so ds waits overlap) -> 8 back-to-back WMMAs.
// B tile (64n x 32k) staged once per block via GLOBAL_LOAD_ASYNC_TO_LDS_B128,
// double-buffered; s_wait_asynccnt 0x1 + barrier covers the current stage.
// EPI: 0=store f32, 1=sigmoid f32, 2=relu^2 -> bf16,
//      3=out=add0+aux+acc (f32), 4=out=add0+aux*acc (f32)
// ---------------------------------------------------------------------------
template <int EPI>
__global__ void k_gemm_wmma(const unsigned short* __restrict__ A,
                            const unsigned short* __restrict__ Bt,
                            int M, int N, int K,
                            const float* __restrict__ add0,
                            const float* __restrict__ aux,
                            void* __restrict__ outv) {
  __shared__ unsigned char s_b[2][64 * LDS_PITCH];

  const int tid  = threadIdx.x;
  const int lane = tid & 31;
  const int wave = tid >> 5;
  const int m0 = blockIdx.y * 256 + wave * 32;  // 32 rows per wave
  const int n0 = blockIdx.x * 64;
  const int koff = (lane >> 4) * 8;

  // staging assignment: thread -> (local n row, 16B chunk within 64B of k-data)
  const int srow = tid >> 2;  // 0..63
  const int schk = tid & 3;   // 0..3
  const unsigned short* bsrc = Bt + (long long)(n0 + srow) * K + schk * 8;
  const unsigned sdst_off = (unsigned)(srow * LDS_PITCH + schk * 16);

  v8f acc[2][4] = {};
  const unsigned short* arow0 = A + (long long)(m0 + (lane & 15)) * K;
  const unsigned short* arow1 = arow0 + (long long)16 * K;
  const int nsteps = K >> 5;

  // prologue: stage B k-step 0; prefetch A k-step 0 into registers
  async_b128_to_lds(lds_off(&s_b[0][0]) + sdst_off, bsrc);
  v16bf a0 = frag_from_global(arow0, 0, koff);
  v16bf a1 = frag_from_global(arow1, 0, koff);

  for (int ks = 0; ks < nsteps; ++ks) {
    const int cur = ks & 1;
    if (ks + 1 < nsteps) {
      async_b128_to_lds(lds_off(&s_b[cur ^ 1][0]) + sdst_off,
                        bsrc + (long long)(ks + 1) * 32);
      asm volatile("s_wait_asynccnt 0x1" ::: "memory");
    } else {
      asm volatile("s_wait_asynccnt 0x0" ::: "memory");
    }
    __syncthreads();  // all waves' async chunks visible

    // prefetch next A fragments (latency hidden behind this step's WMMAs)
    v16bf na0 = a0, na1 = a1;
    if (ks + 1 < nsteps) {
      na0 = frag_from_global(arow0, (ks + 1) * 32, koff);
      na1 = frag_from_global(arow1, (ks + 1) * 32, koff);
    }

    // load all 4 B fragments into distinct registers, then burst 8 WMMAs
    v16bf bfrag[4];
#pragma unroll
    for (int j = 0; j < 4; ++j)
      bfrag[j] = frag_from_lds(&s_b[cur][(16 * j + (lane & 15)) * LDS_PITCH],
                               koff);
#pragma unroll
    for (int j = 0; j < 4; ++j)
      acc[0][j] = __builtin_amdgcn_wmma_f32_16x16x32_bf16(
          false, a0, false, bfrag[j], (short)0, acc[0][j], false, false);
#pragma unroll
    for (int j = 0; j < 4; ++j)
      acc[1][j] = __builtin_amdgcn_wmma_f32_16x16x32_bf16(
          false, a1, false, bfrag[j], (short)0, acc[1][j], false, false);

    __syncthreads();  // done reading 'cur' before it is overwritten next iter
    a0 = na0; a1 = na1;
  }

  // Epilogue. C layout: VGPR v / lane l -> m = msub + v + 8*(l>>4), n = n0+16j+(l&15)
#pragma unroll
  for (int half = 0; half < 2; ++half) {
    const int mbase = m0 + 16 * half + 8 * (lane >> 4);
#pragma unroll
    for (int j = 0; j < 4; ++j) {
      const int n = n0 + 16 * j + (lane & 15);
#pragma unroll
      for (int v = 0; v < 8; ++v) {
        const int m = mbase + v;
        const long long idx = (long long)m * N + n;
        const float x = acc[half][j][v];
        if (EPI == 0) {
          ((float*)outv)[idx] = x;
        } else if (EPI == 1) {
          ((float*)outv)[idx] = 1.f / (1.f + __expf(-x));
        } else if (EPI == 2) {
          const float t = fmaxf(x, 0.f);
          ((unsigned short*)outv)[idx] = f32_to_bf16(t * t);
        } else if (EPI == 3) {
          ((float*)outv)[idx] = add0[idx] + aux[idx] + x;
        } else {
          ((float*)outv)[idx] = add0[idx] + aux[idx] * x;
        }
      }
    }
  }
}

// ---------------------------------------------------------------------------
extern "C" void kernel_launch(void* const* d_in, const int* in_sizes, int n_in,
                              void* d_out, int out_size, void* d_ws, size_t ws_size,
                              hipStream_t stream) {
  (void)in_sizes; (void)n_in; (void)out_size; (void)ws_size;

  const float* hidden       = (const float*)d_in[0];
  const float* att_sx       = (const float*)d_in[1];
  const float* att_aa       = (const float*)d_in[2];
  const float* att_bb       = (const float*)d_in[3];
  const float* att_pp       = (const float*)d_in[4];
  const float* ffd_sx       = (const float*)d_in[5];
  const float* ln1_scale    = (const float*)d_in[6];
  const float* ln1_bias     = (const float*)d_in[7];
  const float* ln2_scale    = (const float*)d_in[8];
  const float* ln2_bias     = (const float*)d_in[9];
  const float* time_decay   = (const float*)d_in[10];
  const float* time_first   = (const float*)d_in[11];
  const float* tm_key       = (const float*)d_in[12];
  const float* tm_value     = (const float*)d_in[13];
  const float* tm_recep     = (const float*)d_in[14];
  const float* Wk           = (const float*)d_in[15];
  const float* Wv           = (const float*)d_in[16];
  const float* Wr           = (const float*)d_in[17];
  const float* Wo           = (const float*)d_in[18];
  const float* ffn_tm_key   = (const float*)d_in[19];
  const float* ffn_tm_recep = (const float*)d_in[20];
  const float* Wfk          = (const float*)d_in[21];
  const float* Wfr          = (const float*)d_in[22];
  const float* Wfv          = (const float*)d_in[23];

  const size_t TH = (size_t)T_SEQ * H_DIM;  // 8M
  const size_t TI = (size_t)T_SEQ * I_DIM;  // 32M
  const size_t HH = (size_t)H_DIM * H_DIM;  // 4M
  const size_t HI = (size_t)H_DIM * I_DIM;  // 16M

  // Workspace layout
  char* w = (char*)d_ws;
  float* x1      = (float*)w; w += TH * 4;
  float* x2      = (float*)w; w += TH * 4;
  float* hidden2 = (float*)w; w += TH * 4;
  float* kbuf    = (float*)w; w += TH * 4;
  float* vbuf    = (float*)w; w += TH * 4;
  float* rbuf    = (float*)w; w += TH * 4;
  float* rrbuf   = (float*)w; w += TH * 4;
  unsigned short* kxb  = (unsigned short*)w; w += TH * 2;
  unsigned short* vxb  = (unsigned short*)w; w += TH * 2;
  unsigned short* rxb  = (unsigned short*)w; w += TH * 2;
  unsigned short* fkb  = (unsigned short*)w; w += TH * 2;
  unsigned short* frb  = (unsigned short*)w; w += TH * 2;
  unsigned short* rcxb = (unsigned short*)w; w += TH * 2;
  unsigned short* kk2b = (unsigned short*)w; w += TI * 2;
  unsigned short* WkT  = (unsigned short*)w; w += HH * 2;
  unsigned short* WvT  = (unsigned short*)w; w += HH * 2;
  unsigned short* WrT  = (unsigned short*)w; w += HH * 2;
  unsigned short* WoT  = (unsigned short*)w; w += HH * 2;
  unsigned short* WfrT = (unsigned short*)w; w += HH * 2;
  unsigned short* WfkT = (unsigned short*)w; w += HI * 2;
  unsigned short* WfvT = (unsigned short*)w; w += HI * 2;

  // Output layout: out[T,H], x1_last[H], aa[H], bb[H], pp[H], x2_last[H]
  float* out       = (float*)d_out;
  float* x1_last_o = out + TH;
  float* aa_o      = out + TH + (size_t)H_DIM;
  float* bb_o      = out + TH + (size_t)2 * H_DIM;
  float* pp_o      = out + TH + (size_t)3 * H_DIM;
  float* x2_last_o = out + TH + (size_t)4 * H_DIM;

  const int BT = 256;
  const int gHH = (int)((HH + BT - 1) / BT);
  const int gHI = (int)((HI + BT - 1) / BT);
  const int gTH = (int)((TH + BT - 1) / BT);

  // 1) weight convert + transpose to bf16 [N][K]
  k_convert_transpose<<<gHH, BT, 0, stream>>>(Wk,  WkT,  H_DIM, H_DIM);
  k_convert_transpose<<<gHH, BT, 0, stream>>>(Wv,  WvT,  H_DIM, H_DIM);
  k_convert_transpose<<<gHH, BT, 0, stream>>>(Wr,  WrT,  H_DIM, H_DIM);
  k_convert_transpose<<<gHH, BT, 0, stream>>>(Wo,  WoT,  H_DIM, H_DIM);
  k_convert_transpose<<<gHH, BT, 0, stream>>>(Wfr, WfrT, H_DIM, H_DIM);
  k_convert_transpose<<<gHI, BT, 0, stream>>>(Wfk, WfkT, H_DIM, I_DIM);
  k_convert_transpose<<<gHI, BT, 0, stream>>>(Wfv, WfvT, I_DIM, H_DIM);

  // 2) LN1 + attention token-shift mix
  k_layernorm<<<T_SEQ, BT, 0, stream>>>(hidden, ln1_scale, ln1_bias, x1);
  k_mix_att<<<gTH, BT, 0, stream>>>(x1, att_sx, tm_key, tm_value, tm_recep,
                                    kxb, vxb, rxb, x1_last_o);

  // 3) k / v / r projections (WMMA)
  dim3 blk(256, 1, 1);
  dim3 grdH(H_DIM / 64, T_SEQ / 256, 1);
  dim3 grdI(I_DIM / 64, T_SEQ / 256, 1);
  k_gemm_wmma<0><<<grdH, blk, 0, stream>>>(kxb, WkT, T_SEQ, H_DIM, H_DIM,
                                           nullptr, nullptr, kbuf);
  k_gemm_wmma<0><<<grdH, blk, 0, stream>>>(vxb, WvT, T_SEQ, H_DIM, H_DIM,
                                           nullptr, nullptr, vbuf);
  k_gemm_wmma<1><<<grdH, blk, 0, stream>>>(rxb, WrT, T_SEQ, H_DIM, H_DIM,
                                           nullptr, nullptr, rbuf);

  // 4) WKV recurrence (emits bf16 r*cx + final aa/bb/pp)
  k_wkv<<<H_DIM / BT, BT, 0, stream>>>(kbuf, vbuf, rbuf, att_aa, att_bb, att_pp,
                                       time_decay, time_first, rcxb,
                                       aa_o, bb_o, pp_o);

  // 5) output projection, fused residual: hidden2 = hidden + x1 + (r*cx)@Wo
  k_gemm_wmma<3><<<grdH, blk, 0, stream>>>(rcxb, WoT, T_SEQ, H_DIM, H_DIM,
                                           hidden, x1, hidden2);

  // 6) LN2 + FFN token-shift mix
  k_layernorm<<<T_SEQ, BT, 0, stream>>>(hidden2, ln2_scale, ln2_bias, x2);
  k_mix_ffn<<<gTH, BT, 0, stream>>>(x2, ffd_sx, ffn_tm_key, ffn_tm_recep,
                                    fkb, frb, x2_last_o);

  // 7) FFN: rr = sigmoid(fr@Wfr); kk2 = relu(fk@Wfk)^2 (bf16);
  //    out = hidden2 + rr * (kk2@Wfv)
  k_gemm_wmma<1><<<grdH, blk, 0, stream>>>(frb, WfrT, T_SEQ, H_DIM, H_DIM,
                                           nullptr, nullptr, rrbuf);
  k_gemm_wmma<2><<<grdI, blk, 0, stream>>>(fkb, WfkT, T_SEQ, I_DIM, H_DIM,
                                           nullptr, nullptr, kk2b);
  k_gemm_wmma<4><<<grdH, blk, 0, stream>>>(kk2b, WfvT, T_SEQ, H_DIM, I_DIM,
                                           hidden2, rrbuf, out);
}
